// MSAWithRoPE_77816217469182
// MI455X (gfx1250) — matrix operand order repaired and checked by reading, not compiled, not run on hardware
//
#include <hip/hip_runtime.h>

#define B_  8
#define C_  384
#define H_  48
#define W_  48
#define NH  6
#define HD  64
#define N_  (H_*W_)      // 2304
#define EPS 1e-6f

typedef __attribute__((ext_vector_type(16))) _Float16 v16h;
typedef __attribute__((ext_vector_type(8)))  _Float16 v8h;
typedef __attribute__((ext_vector_type(8)))  float    v8f;
typedef int v4i __attribute__((__vector_size__(16)));   // matches builtin param type

union V16H { v16h v; v8h h[2]; _Float16 e[16]; };

// ---- CDNA5 async global->LDS path (guarded; VGPR-staging fallback) --------
#if defined(__has_builtin)
#  if __has_builtin(__builtin_amdgcn_global_load_async_to_lds_b128)
#    define USE_ASYNC_LDS 1
#  endif
#endif
#ifndef USE_ASYNC_LDS
#  define USE_ASYNC_LDS 0
#endif

typedef __attribute__((address_space(1))) v4i* gv4i_p;   // global (AS1)
typedef __attribute__((address_space(3))) v4i* lv4i_p;   // LDS (AS3)

static __device__ __forceinline__ void async_cp16(const void* g, void* l) {
#if USE_ASYNC_LDS
    __builtin_amdgcn_global_load_async_to_lds_b128(
        (gv4i_p)(uintptr_t)g, (lv4i_p)(uint32_t)(uintptr_t)l, 0, 0);
#else
    *(v8h*)l = *(const v8h*)g;
#endif
}

static __device__ __forceinline__ void wait_async() {
#if USE_ASYNC_LDS
#  if __has_builtin(__builtin_amdgcn_s_wait_asynccnt)
    __builtin_amdgcn_s_wait_asynccnt(0);
#  else
    asm volatile("s_wait_asynccnt 0x0" ::: "memory");
#  endif
#endif
}

static __device__ __forceinline__ float fast_exp2(float x) {
#if defined(__has_builtin)
#  if __has_builtin(__builtin_amdgcn_exp2f)
    return __builtin_amdgcn_exp2f(x);
#  else
    return exp2f(x);
#  endif
#else
    return exp2f(x);
#endif
}

static __device__ __forceinline__ v8f zero8() {
    v8f z;
#pragma unroll
    for (int r = 0; r < 8; ++r) z[r] = 0.0f;
    return z;
}

#define LOG2E 1.44269504088896f

// ---------------------------------------------------------------------------
// Kernel 1: depthwise 3x3 circular conv (grouped oc -> ic = oc/3), RMS norm,
// RoPE-Mixed. Q is pre-scaled by hd^-0.5 * log2(e) so attention softmax can
// use raw v_exp_f32 (base-2). Writes:
//   Qh, Kh : (B*NH, N, 64) f16   (hd contiguous)
//   Vt     : (B*NH, 64, N) f16   (transposed: keys contiguous per d-row)
// ---------------------------------------------------------------------------
__global__ __launch_bounds__(256) void prep_kernel(
    const float* __restrict__ x, const float* __restrict__ conv_w,
    const float* __restrict__ qnw, const float* __restrict__ knw,
    const float* __restrict__ freqs,
    _Float16* __restrict__ Qh, _Float16* __restrict__ Kh,
    _Float16* __restrict__ Vt)
{
    int tid = blockIdx.x * blockDim.x + threadIdx.x;
    if (tid >= B_ * NH * N_) return;
    int n    = tid % N_;
    int bh   = tid / N_;
    int head = bh % NH;
    int b    = bh / NH;
    int y = n / W_, xc = n % W_;
    int ym = (y + H_ - 1) % H_, yp = (y + 1) % H_;
    int xm = (xc + W_ - 1) % W_, xp2 = (xc + 1) % W_;
    int ys[3] = {ym, y, yp};
    int xs[3] = {xm, xc, xp2};

    float cs[HD/2], sn[HD/2];
#pragma unroll
    for (int i = 0; i < HD/2; ++i) {
        float ang = (float)y * freqs[2*i] + (float)xc * freqs[2*i + 1];
        __sincosf(ang, &sn[i], &cs[i]);
    }

    float vec[HD];
    for (int which = 0; which < 3; ++which) {
        int jbase = head * HD;
        for (int d = 0; d < HD; ++d) {
            int oc = which * C_ + jbase + d;      // weight row in (3C,1,3,3)
            int ic = oc / 3;                      // grouped conv input channel
            const float* wp = conv_w + oc * 9;
            const float* xp = x + ((size_t)b * C_ + ic) * N_;
            float acc = 0.f;
#pragma unroll
            for (int dy = 0; dy < 3; ++dy)
#pragma unroll
                for (int dx = 0; dx < 3; ++dx)
                    acc += xp[ys[dy] * W_ + xs[dx]] * wp[dy * 3 + dx];
            vec[d] = acc;
        }
        if (which < 2) {
            const float* nw = (which == 0) ? qnw : knw;
            float ss = 0.f;
#pragma unroll
            for (int d = 0; d < HD; ++d) ss += vec[d] * vec[d];
            float r = rsqrtf(ss * (1.0f / HD) + EPS);
#pragma unroll
            for (int d = 0; d < HD; ++d) vec[d] = vec[d] * r * nw[d];
#pragma unroll
            for (int i = 0; i < HD/2; ++i) {      // RoPE
                float a = vec[2*i], bb = vec[2*i + 1];
                vec[2*i]     = a * cs[i] - bb * sn[i];
                vec[2*i + 1] = bb * cs[i] + a * sn[i];
            }
            float scale = (which == 0) ? (0.125f * LOG2E) : 1.0f;
            _Float16* dst = ((which == 0) ? Qh : Kh) + ((size_t)bh * N_ + n) * HD;
#pragma unroll
            for (int d = 0; d < HD; ++d) dst[d] = (_Float16)(vec[d] * scale);
        } else {
#pragma unroll
            for (int d = 0; d < HD; ++d)
                Vt[((size_t)bh * HD + d) * N_ + n] = (_Float16)vec[d];
        }
    }
}

// ---------------------------------------------------------------------------
// Kernel 2: flash attention via WMMA (S^T = K Q^T, O^T = V^T P^T), with
// double-buffered async global->LDS staging of K/V tiles (ASYNCcnt pipe).
// Block: 128 thr = 4 waves; wave owns 16 queries; key tile = 32.
// ---------------------------------------------------------------------------
__global__ __launch_bounds__(128) void attn_kernel(
    const float* __restrict__ x,
    const _Float16* __restrict__ Qh, const _Float16* __restrict__ Kh,
    const _Float16* __restrict__ Vt, float* __restrict__ out)
{
    __shared__ __align__(16) _Float16 KT[2][32 * HD];  // [buf][key][d]
    __shared__ __align__(16) _Float16 VT[2][HD * 32];  // [buf][d][key]

    const int bh   = blockIdx.x;
    const int qblk = blockIdx.y;
    const int tid  = threadIdx.x;
    const int wave = tid >> 5;
    const int lane = tid & 31;
    const int lhalf = lane & 15;
    const bool hi  = lane >= 16;

    const int q = qblk * 64 + wave * 16 + lhalf;   // this lane's query index

    const _Float16* Qp = Qh + (size_t)bh * N_ * HD;
    const _Float16* Kp = Kh + (size_t)bh * N_ * HD;
    const _Float16* Vp = Vt + (size_t)bh * HD * N_;

    // Q^T B-operands (32x16), held for the whole loop.
    // B layout: lanes 0-15 hold contraction K=0..7 & 16..23, lanes 16-31 K=8..15 & 24..31.
    V16H qb0, qb1;
    {
        const _Float16* qrow = Qp + (size_t)q * HD;
        int d0 = hi ? 8 : 0;
        qb0.h[0] = *(const v8h*)(qrow + d0);
        qb0.h[1] = *(const v8h*)(qrow + d0 + 16);
        qb1.h[0] = *(const v8h*)(qrow + 32 + d0);
        qb1.h[1] = *(const v8h*)(qrow + 32 + d0 + 16);
    }

    // issue async copies for one 32-key tile (K: 32x64, V^T: 64x32) into buf
    auto stage = [&](int buf, int k0) {
#pragma unroll
        for (int it = 0; it < 2; ++it) {
            int idx = tid + it * 128;                  // 256 x 16B per tile
            int kr = idx >> 3, kc = idx & 7;
            async_cp16(Kp + (size_t)(k0 + kr) * HD + kc * 8,
                       &KT[buf][kr * HD + kc * 8]);
            int vr = idx >> 2, vc = idx & 3;
            async_cp16(Vp + (size_t)vr * N_ + k0 + vc * 8,
                       &VT[buf][vr * 32 + vc * 8]);
        }
    };

    v8f ot0 = zero8(), ot1 = zero8(), ot2 = zero8(), ot3 = zero8();
    float m = -1e30f, l = 0.f;

    const int kiters = N_ / 32;
    stage(0, 0);
    for (int kt = 0; kt < kiters; ++kt) {
        const int cur = kt & 1;
        wait_async();          // my copies into KT/VT[cur] complete
        __syncthreads();       // everyone's copies complete / prior reads done
        if (kt + 1 < kiters) stage(cur ^ 1, (kt + 1) * 32);

        const _Float16* Ktile = &KT[cur][0];
        const _Float16* Vtile = &VT[cur][0];

        // ---- S^T = K_tile (keys x 64) * Q^T (64 x 16q): 4 WMMAs ----------
        // A layout 16x32: lanes 0-15: M=row, K=0..7 & 16..23; lanes 16-31: K=8..15 & 24..31
        const int doff = hi ? 8 : 0;
        v8f st0 = zero8(), st1 = zero8();
        {
            V16H a;
            int krow = lhalf;                       // keys 0..15
            a.h[0] = *(const v8h*)(Ktile + krow * HD + doff);
            a.h[1] = *(const v8h*)(Ktile + krow * HD + 16 + doff);
            st0 = __builtin_amdgcn_wmma_f32_16x16x32_f16(false, a.v, false, qb0.v, (short)0, st0, false, false);
            a.h[0] = *(const v8h*)(Ktile + krow * HD + 32 + doff);
            a.h[1] = *(const v8h*)(Ktile + krow * HD + 48 + doff);
            st0 = __builtin_amdgcn_wmma_f32_16x16x32_f16(false, a.v, false, qb1.v, (short)0, st0, false, false);
            krow = 16 + lhalf;                      // keys 16..31
            a.h[0] = *(const v8h*)(Ktile + krow * HD + doff);
            a.h[1] = *(const v8h*)(Ktile + krow * HD + 16 + doff);
            st1 = __builtin_amdgcn_wmma_f32_16x16x32_f16(false, a.v, false, qb0.v, (short)0, st1, false, false);
            a.h[0] = *(const v8h*)(Ktile + krow * HD + 32 + doff);
            a.h[1] = *(const v8h*)(Ktile + krow * HD + 48 + doff);
            st1 = __builtin_amdgcn_wmma_f32_16x16x32_f16(false, a.v, false, qb1.v, (short)0, st1, false, false);
        }

        // ---- online softmax (base-2 domain; log2e folded into Q) ----------
        float rmax = -1e30f;
#pragma unroll
        for (int r = 0; r < 8; ++r) { rmax = fmaxf(rmax, st0[r]); rmax = fmaxf(rmax, st1[r]); }
        rmax = fmaxf(rmax, __shfl_xor(rmax, 16, 32));
        float m_new = fmaxf(m, rmax);
        float alpha = fast_exp2(m - m_new);
        float p0[8], p1[8], lsum = 0.f;
#pragma unroll
        for (int r = 0; r < 8; ++r) {
            p0[r] = fast_exp2(st0[r] - m_new);
            p1[r] = fast_exp2(st1[r] - m_new);
            lsum += p0[r] + p1[r];
        }
        lsum += __shfl_xor(lsum, 16, 32);
        l = l * alpha + lsum;
        m = m_new;
#pragma unroll
        for (int r = 0; r < 8; ++r) {
            ot0[r] *= alpha; ot1[r] *= alpha; ot2[r] *= alpha; ot3[r] *= alpha;
        }

        // ---- pack P^T (32 keys x 16 q) into B layout ----------------------
        // S^T C-layout: lanes<16 hold keys {0..7} (st0) & {16..23} (st1);
        //               lanes>=16 hold keys {8..15} & {24..31}.
        V16H pb;
#pragma unroll
        for (int r = 0; r < 8; ++r) {
            pb.e[r]     = (_Float16)p0[r];   // K = 8h+0..7
            pb.e[8 + r] = (_Float16)p1[r];   // K = 16+8h+0..7
        }

        // ---- O^T += V^T (64 x 32keys) * P^T (32keys x 16q): 4 WMMAs -------
        {
            V16H va;
            int drow = lhalf;
            va.h[0] = *(const v8h*)(Vtile + drow * 32 + doff);
            va.h[1] = *(const v8h*)(Vtile + drow * 32 + 16 + doff);
            ot0 = __builtin_amdgcn_wmma_f32_16x16x32_f16(false, va.v, false, pb.v, (short)0, ot0, false, false);
            drow = 16 + lhalf;
            va.h[0] = *(const v8h*)(Vtile + drow * 32 + doff);
            va.h[1] = *(const v8h*)(Vtile + drow * 32 + 16 + doff);
            ot1 = __builtin_amdgcn_wmma_f32_16x16x32_f16(false, va.v, false, pb.v, (short)0, ot1, false, false);
            drow = 32 + lhalf;
            va.h[0] = *(const v8h*)(Vtile + drow * 32 + doff);
            va.h[1] = *(const v8h*)(Vtile + drow * 32 + 16 + doff);
            ot2 = __builtin_amdgcn_wmma_f32_16x16x32_f16(false, va.v, false, pb.v, (short)0, ot2, false, false);
            drow = 48 + lhalf;
            va.h[0] = *(const v8h*)(Vtile + drow * 32 + doff);
            va.h[1] = *(const v8h*)(Vtile + drow * 32 + 16 + doff);
            ot3 = __builtin_amdgcn_wmma_f32_16x16x32_f16(false, va.v, false, pb.v, (short)0, ot3, false, false);
        }
    }

    // ---- epilogue: normalize, residual add, scatter O^T -> (B,C,H,W) ------
    const float inv_l = 1.0f / l;
    const int head = bh % NH, b = bh / NH;
    const int dbase = hi ? 8 : 0;
    const size_t base = ((size_t)b * C_ + head * HD) * N_;
#pragma unroll
    for (int r = 0; r < 8; ++r) {
        size_t i0 = base + (size_t)(dbase + r) * N_ + q;
        out[i0] = x[i0] + ot0[r] * inv_l;
        size_t i1 = base + (size_t)(16 + dbase + r) * N_ + q;
        out[i1] = x[i1] + ot1[r] * inv_l;
        size_t i2 = base + (size_t)(32 + dbase + r) * N_ + q;
        out[i2] = x[i2] + ot2[r] * inv_l;
        size_t i3 = base + (size_t)(48 + dbase + r) * N_ + q;
        out[i3] = x[i3] + ot3[r] * inv_l;
    }
}

extern "C" void kernel_launch(void* const* d_in, const int* in_sizes, int n_in,
                              void* d_out, int out_size, void* d_ws, size_t ws_size,
                              hipStream_t stream) {
    const float* x      = (const float*)d_in[0];
    const float* conv_w = (const float*)d_in[1];
    const float* qnw    = (const float*)d_in[2];
    const float* knw    = (const float*)d_in[3];
    const float* freqs  = (const float*)d_in[4];
    float* out = (float*)d_out;

    const size_t per = (size_t)B_ * NH * N_ * HD;   // f16 elements per buffer
    _Float16* Qh = (_Float16*)d_ws;
    _Float16* Kh = Qh + per;
    _Float16* Vt = Kh + per;

    int total = B_ * NH * N_;
    prep_kernel<<<(total + 255) / 256, 256, 0, stream>>>(x, conv_w, qnw, knw, freqs, Qh, Kh, Vt);

    dim3 grid(B_ * NH, N_ / 64);
    attn_kernel<<<grid, 128, 0, stream>>>(x, Qh, Kh, Vt, out);
}
